// GNN_M_graphpred_86646670229663
// MI455X (gfx1250) — matrix operand order repaired and checked by reading, not compiled, not run on hardware
//
#include <hip/hip_runtime.h>
#include <hip/hip_bf16.h>

// ---------------------------------------------------------------------------
// GIN graph-pred forward for MI455X (gfx1250, wave32, WMMA).
// GEMMs: v_wmma_f32_16x16x32_bf16, 128x128 block tile, padded operands so the
// staging path is branch-free b128 loads/stores. Scatter-add & pooling use
// hardware global_atomic_add_f32; BN stats use ds_add_f32 in LDS.
// ---------------------------------------------------------------------------

#define N_NODES   100000
#define N_EDGES   200000
#define EMB       300
#define N_LAYERS  5
#define N_GRAPHS  2000
#define N_TASKS   12
#define BN_EPS    1e-5f

// padded geometry
#define M_PAD     100096          // 782 * 128
#define K1_PAD    320             // EMB padded to 32
#define N1_PAD    640             // 2*EMB padded to 128
#define K2_PAD    640             // == N1_PAD
#define N2_PAD    384             // EMB padded to 128

typedef __attribute__((ext_vector_type(16))) __bf16 v16bf;
typedef __attribute__((ext_vector_type(8)))  __bf16 v8bf;
typedef __attribute__((ext_vector_type(8)))  float  v8f;

__device__ __forceinline__ void atomAddGlobal(float* p, float v) {
    (void)__hip_atomic_fetch_add(p, v, __ATOMIC_RELAXED, __HIP_MEMORY_SCOPE_AGENT);
}
__device__ __forceinline__ void atomAddLds(float* p, float v) {
    (void)__hip_atomic_fetch_add(p, v, __ATOMIC_RELAXED, __HIP_MEMORY_SCOPE_WORKGROUP);
}

// --------------------------- elementwise kernels ---------------------------

__global__ __launch_bounds__(256) void k_zero(float* p, int n) {
    int i = blockIdx.x * 256 + threadIdx.x;
    if (i < n) p[i] = 0.0f;
}

// strided pad-cast: out[r][c] = (r<inR && c<inC) ? bf16(in[r][c]) : 0
__global__ __launch_bounds__(256) void k_pad_cast(const float* __restrict__ in,
                                                  __bf16* __restrict__ out,
                                                  int inR, int inC,
                                                  int outR, int outC) {
    long long i = (long long)blockIdx.x * 256 + threadIdx.x;
    long long total = (long long)outR * outC;
    if (i >= total) return;
    int r = (int)(i / outC), c = (int)(i - (long long)r * outC);
    out[i] = (r < inR && c < inC) ? (__bf16)in[(size_t)r * inC + c] : (__bf16)0.0f;
}

// padded float copy (for bias vectors)
__global__ __launch_bounds__(256) void k_pad_f32(const float* __restrict__ in,
                                                 float* __restrict__ out,
                                                 int n, int npad) {
    int i = blockIdx.x * 256 + threadIdx.x;
    if (i < npad) out[i] = (i < n) ? in[i] : 0.0f;
}

// h[n][d] = x_emb1[x[n,0]][d] + x_emb2[x[n,1]][d]
__global__ __launch_bounds__(256) void k_embed(const int* __restrict__ x,
                                               const float* __restrict__ e1,
                                               const float* __restrict__ e2,
                                               float* __restrict__ h, int total) {
    int i = blockIdx.x * 256 + threadIdx.x;
    if (i >= total) return;
    int r = i / EMB, d = i - r * EMB;
    h[i] = e1[x[2 * r] * EMB + d] + e2[x[2 * r + 1] * EMB + d];
}

// agg[n][d] = h[n][d] + e1_self[d] + e2_self[d]   (self-loop init)
__global__ __launch_bounds__(256) void k_selfloop(const float* __restrict__ h,
                                                  const float* __restrict__ e1sl,
                                                  const float* __restrict__ e2sl,
                                                  float* __restrict__ agg, int total) {
    int i = blockIdx.x * 256 + threadIdx.x;
    if (i >= total) return;
    int d = i % EMB;
    agg[i] = h[i] + e1sl[d] + e2sl[d];
}

// agg[col[e]][d] += h[row[e]][d] + e1[ea0[e]][d] + e2[ea1[e]][d]
__global__ __launch_bounds__(256) void k_edge_scatter(const int* __restrict__ ei,
                                                      const int* __restrict__ ea,
                                                      const float* __restrict__ h,
                                                      const float* __restrict__ e1,
                                                      const float* __restrict__ e2,
                                                      float* __restrict__ agg, int total) {
    int i = blockIdx.x * 256 + threadIdx.x;
    if (i >= total) return;
    int e = i / EMB, d = i - e * EMB;
    int r  = ei[e];
    int c  = ei[N_EDGES + e];
    int a0 = ea[2 * e];
    int a1 = ea[2 * e + 1];
    float v = h[(size_t)r * EMB + d] + e1[a0 * EMB + d] + e2[a1 * EMB + d];
    atomAddGlobal(&agg[(size_t)c * EMB + d], v);
}

// ------------------------------ WMMA GEMM ----------------------------------
// C[M,N] = A[M,K]x B[K,N] + bias (bf16 in, f32 acc). Block tile 128x128,
// BK=32; 8 waves in a 4(m) x 2(n) grid, each wave owns 32x64 (2x4 WMMA tiles).
// A/B are padded in workspace: NO bounds checks in the staging path; stores
// masked only in the epilogue. Register-level prefetch of the next k-tile.

template <bool RELU, bool OUT_BF16>
__global__ __launch_bounds__(256) void k_gemm_wmma(const __bf16* __restrict__ A,
                                                   const __bf16* __restrict__ B,
                                                   const float* __restrict__ bias,
                                                   float* __restrict__ outF,
                                                   __bf16* __restrict__ outB,
                                                   int Mreal, int Nreal, int ldc,
                                                   int lda, int ldb, int Kpad) {
    __shared__ alignas(16) __bf16 ldsA[128 * 32];   // 8 KB
    __shared__ alignas(16) __bf16 ldsB[32 * 128];   // 8 KB

    const int tid  = threadIdx.x;
    const int lane = tid & 31;
    const int wave = tid >> 5;
    const int wm   = (wave & 3) * 32;   // wave row base in block tile
    const int wn   = (wave >> 2) * 64;  // wave col base in block tile
    const int blockM = blockIdx.x * 128;
    const int blockN = blockIdx.y * 128;

    // staging assignments (all unconditional, aligned)
    const int rowA  = tid >> 1;             // 0..127
    const int halfA = (tid & 1) << 4;       // 0 / 16
    const int kRowB = tid >> 3;             // 0..31
    const int segB  = (tid & 7) << 4;       // 0..112

    const __bf16* aSrc = A + (size_t)(blockM + rowA) * lda + halfA;
    const __bf16* bSrc = B + (size_t)kRowB * ldb + blockN + segB;

    v8f acc[2][4] = {};

    const int kSteps = Kpad >> 5;
    uint4 ra0 = ((const uint4*)aSrc)[0];
    uint4 ra1 = ((const uint4*)aSrc)[1];
    uint4 rb0 = ((const uint4*)bSrc)[0];
    uint4 rb1 = ((const uint4*)bSrc)[1];

    for (int ks = 0; ks < kSteps; ++ks) {
        // commit staged registers to LDS
        {
            uint4* dA = (uint4*)&ldsA[rowA * 32 + halfA];
            dA[0] = ra0;   // halfA..halfA+7
            // second uint4 goes 8 elements further
            *(uint4*)&ldsA[rowA * 32 + halfA + 8] = ra1;
            uint4* dB = (uint4*)&ldsB[kRowB * 128 + segB];
            dB[0] = rb0;
            *(uint4*)&ldsB[kRowB * 128 + segB + 8] = rb1;
        }
        __syncthreads();

        // prefetch next k-tile while WMMAs run
        if (ks + 1 < kSteps) {
            aSrc += 32;
            bSrc += (size_t)32 * ldb;
            ra0 = ((const uint4*)aSrc)[0];
            ra1 = ((const uint4*)aSrc)[1];
            rb0 = ((const uint4*)bSrc)[0];
            rb1 = ((const uint4*)bSrc)[1];
        }

        // fragments (layouts per CDNA5 ISA §7.12.2)
        const int r0 = lane & 15;
        const int kb = (lane >> 4) << 3;    // 0 / 8
        v16bf aF[2];
        #pragma unroll
        for (int tm = 0; tm < 2; ++tm) {
            const __bf16* ap = &ldsA[(wm + tm * 16 + r0) * 32 + kb];
            v8bf lo = *(const v8bf*)ap;          // k = kb .. kb+7
            v8bf hi = *(const v8bf*)(ap + 16);   // k = kb+16 .. kb+23
            aF[tm] = __builtin_shufflevector(lo, hi, 0, 1, 2, 3, 4, 5, 6, 7,
                                                     8, 9, 10, 11, 12, 13, 14, 15);
        }
        v16bf bF[4];
        #pragma unroll
        for (int tn = 0; tn < 4; ++tn) {
            const __bf16* bq = &ldsB[wn + tn * 16 + r0]; // column n = lane%16
            #pragma unroll
            for (int s = 0; s < 16; ++s)
                bF[tn][s] = bq[((s < 8) ? (kb + s) : (16 + kb + s - 8)) * 128];
        }

        #pragma unroll
        for (int tm = 0; tm < 2; ++tm)
            #pragma unroll
            for (int tn = 0; tn < 4; ++tn)
                acc[tm][tn] = __builtin_amdgcn_wmma_f32_16x16x32_bf16(
                    false, aF[tm], false, bF[tn], (short)0, acc[tm][tn], false, false);

        __syncthreads();
    }

    // epilogue: bias (+ReLU); masked stores with real output geometry
    const int cLo = lane & 15;
    const int rHi = (lane >> 4) << 3;       // 0 / 8
    #pragma unroll
    for (int tm = 0; tm < 2; ++tm) {
        #pragma unroll
        for (int tn = 0; tn < 4; ++tn) {
            const int col = blockN + wn + tn * 16 + cLo;
            if (col >= Nreal) continue;
            const float bv = bias[col];
            #pragma unroll
            for (int v = 0; v < 8; ++v) {
                const int row = blockM + wm + tm * 16 + rHi + v;
                if (row >= Mreal) continue;
                float val = acc[tm][tn][v] + bv;
                if (RELU) val = fmaxf(val, 0.0f);
                const size_t off = (size_t)row * ldc + col;
                if constexpr (OUT_BF16) outB[off] = (__bf16)val;
                else                    outF[off] = val;
            }
        }
    }
}

// ------------------------------ batch norm ---------------------------------

__global__ __launch_bounds__(256) void k_bn_stats(const float* __restrict__ h,
                                                  float* __restrict__ sum,
                                                  float* __restrict__ sq, int n) {
    __shared__ float ls[EMB];
    __shared__ float lq[EMB];
    for (int i = threadIdx.x; i < EMB; i += 256) { ls[i] = 0.0f; lq[i] = 0.0f; }
    __syncthreads();
    const int rowBase = blockIdx.x * 256;
    for (int i = threadIdx.x; i < 256 * EMB; i += 256) {
        int r = rowBase + i / EMB;
        if (r < n) {
            int f = i % EMB;
            float v = h[(size_t)r * EMB + f];
            atomAddLds(&ls[f], v);
            atomAddLds(&lq[f], v * v);
        }
    }
    __syncthreads();
    for (int i = threadIdx.x; i < EMB; i += 256) {
        atomAddGlobal(&sum[i], ls[i]);
        atomAddGlobal(&sq[i], lq[i]);
    }
}

__global__ __launch_bounds__(256) void k_bn_apply(float* __restrict__ h,
                                                  const float* __restrict__ sum,
                                                  const float* __restrict__ sq,
                                                  const float* __restrict__ gamma,
                                                  const float* __restrict__ beta,
                                                  int total, int n, int relu) {
    int i = blockIdx.x * 256 + threadIdx.x;
    if (i >= total) return;
    int f = i % EMB;
    float invN = 1.0f / (float)n;
    float mean = sum[f] * invN;
    float var  = sq[f] * invN - mean * mean;
    float v = (h[i] - mean) * __frsqrt_rn(var + BN_EPS) * gamma[f] + beta[f];
    if (relu) v = fmaxf(v, 0.0f);
    h[i] = v;
}

// ------------------------------- pooling -----------------------------------

__global__ __launch_bounds__(256) void k_pool_accum(const float* __restrict__ h,
                                                    const int* __restrict__ batch,
                                                    float* __restrict__ pool, int total) {
    int i = blockIdx.x * 256 + threadIdx.x;
    if (i >= total) return;
    int r = i / EMB, d = i - r * EMB;
    atomAddGlobal(&pool[(size_t)batch[r] * EMB + d], h[i]);
}

__global__ __launch_bounds__(256) void k_pool_count(const int* __restrict__ batch,
                                                    float* __restrict__ cnt, int n) {
    int i = blockIdx.x * 256 + threadIdx.x;
    if (i < n) atomAddGlobal(&cnt[batch[i]], 1.0f);
}

// out[g][t] = (pool[g]/max(cnt,1)) . Wp[:,t] + bp[t]
__global__ __launch_bounds__(256) void k_head(const float* __restrict__ pool,
                                              const float* __restrict__ cnt,
                                              const float* __restrict__ Wp,
                                              const float* __restrict__ bp,
                                              float* __restrict__ out, int total) {
    int i = blockIdx.x * 256 + threadIdx.x;
    if (i >= total) return;
    int g = i / N_TASKS, t = i - g * N_TASKS;
    float inv = 1.0f / fmaxf(cnt[g], 1.0f);
    float s = bp[t];
    const float* pg = pool + (size_t)g * EMB;
    for (int d = 0; d < EMB; ++d) s += pg[d] * inv * Wp[d * N_TASKS + t];
    out[i] = s;
}

// ------------------------------- launcher ----------------------------------

static inline int nblk(long long total) { return (int)((total + 255) / 256); }

extern "C" void kernel_launch(void* const* d_in, const int* in_sizes, int n_in,
                              void* d_out, int out_size, void* d_ws, size_t ws_size,
                              hipStream_t stream) {
    (void)in_sizes; (void)n_in; (void)out_size; (void)ws_size;

    const int*   x        = (const int*)  d_in[0];
    const int*   eidx     = (const int*)  d_in[1];
    const int*   eattr    = (const int*)  d_in[2];
    const int*   batch    = (const int*)  d_in[3];
    const float* x_emb1   = (const float*)d_in[4];
    const float* x_emb2   = (const float*)d_in[5];
    const float* edge_e1  = (const float*)d_in[6];   // [5][6][300]
    const float* edge_e2  = (const float*)d_in[7];   // [5][3][300]
    const float* W1       = (const float*)d_in[8];   // [5][300][600]
    const float* b1       = (const float*)d_in[9];   // [5][600]
    const float* W2       = (const float*)d_in[10];  // [5][600][300]
    const float* b2       = (const float*)d_in[11];  // [5][300]
    const float* bn_gamma = (const float*)d_in[12];  // [5][300]
    const float* bn_beta  = (const float*)d_in[13];  // [5][300]
    const float* Wp       = (const float*)d_in[14];  // [300][12]
    const float* bp       = (const float*)d_in[15];  // [12]
    float* out = (float*)d_out;

    // workspace carve-up (256B-aligned)
    char* base = (char*)d_ws;
    size_t off = 0;
    auto carve = [&](size_t bytes) -> char* {
        char* p = base + off;
        off = (off + bytes + 255) & ~(size_t)255;
        return p;
    };
    float*  h     = (float*) carve((size_t)N_NODES * EMB * 4);            // 120 MB
    float*  agg   = (float*) carve((size_t)N_NODES * EMB * 4);            // 120 MB
    __bf16* aggb  = (__bf16*)carve((size_t)M_PAD * K1_PAD * 2);           //  64 MB
    __bf16* tmid  = (__bf16*)carve((size_t)M_PAD * N1_PAD * 2);           // 128 MB
    __bf16* w1b   = (__bf16*)carve((size_t)K1_PAD * N1_PAD * 2);
    __bf16* w2b   = (__bf16*)carve((size_t)K2_PAD * N2_PAD * 2);
    float*  b1p   = (float*) carve(N1_PAD * 4);
    float*  b2p   = (float*) carve(N2_PAD * 4);
    float*  bnSum = (float*) carve(EMB * 4);
    float*  bnSq  = (float*) carve(EMB * 4);
    float*  pool  = (float*) carve((size_t)N_GRAPHS * EMB * 4);
    float*  cnt   = (float*) carve((size_t)N_GRAPHS * 4);

    const long long nodeElems = (long long)N_NODES * EMB;   // 30M
    const long long edgeElems = (long long)N_EDGES * EMB;   // 60M

    // h0 = x_emb1[x[:,0]] + x_emb2[x[:,1]]
    k_embed<<<nblk(nodeElems), 256, 0, stream>>>(x, x_emb1, x_emb2, h, (int)nodeElems);

    for (int l = 0; l < N_LAYERS; ++l) {
        const float* e1l = edge_e1 + (size_t)l * 6 * EMB;
        const float* e2l = edge_e2 + (size_t)l * 3 * EMB;
        // self loops: edge_attr = [4, 0]
        k_selfloop<<<nblk(nodeElems), 256, 0, stream>>>(
            h, e1l + 4 * EMB, e2l + 0 * EMB, agg, (int)nodeElems);
        // real edges: agg[col] += h[row] + e1[ea0] + e2[ea1]
        k_edge_scatter<<<nblk(edgeElems), 256, 0, stream>>>(
            eidx, eattr, h, e1l, e2l, agg, (int)edgeElems);

        // padded bf16 copies: activations, weights, biases (pads zero-filled)
        k_pad_cast<<<nblk((long long)M_PAD * K1_PAD), 256, 0, stream>>>(
            agg, aggb, N_NODES, EMB, M_PAD, K1_PAD);
        k_pad_cast<<<nblk((long long)K1_PAD * N1_PAD), 256, 0, stream>>>(
            W1 + (size_t)l * EMB * 2 * EMB, w1b, EMB, 2 * EMB, K1_PAD, N1_PAD);
        k_pad_cast<<<nblk((long long)K2_PAD * N2_PAD), 256, 0, stream>>>(
            W2 + (size_t)l * 2 * EMB * EMB, w2b, 2 * EMB, EMB, K2_PAD, N2_PAD);
        k_pad_f32<<<nblk(N1_PAD), 256, 0, stream>>>(b1 + (size_t)l * 2 * EMB, b1p,
                                                    2 * EMB, N1_PAD);
        k_pad_f32<<<nblk(N2_PAD), 256, 0, stream>>>(b2 + (size_t)l * EMB, b2p,
                                                    EMB, N2_PAD);

        // GIN MLP: tmid = relu(agg @ W1 + b1) [bf16, padded cols exactly 0]
        {
            dim3 grid(M_PAD / 128, N1_PAD / 128);
            k_gemm_wmma<true, true><<<grid, 256, 0, stream>>>(
                aggb, w1b, b1p, nullptr, tmid,
                /*Mreal=*/M_PAD, /*Nreal=*/N1_PAD, /*ldc=*/N1_PAD,
                /*lda=*/K1_PAD, /*ldb=*/N1_PAD, /*Kpad=*/K1_PAD);
        }
        // h = tmid @ W2 + b2  (fp32, real geometry 100000 x 300)
        {
            dim3 grid(M_PAD / 128, N2_PAD / 128);
            k_gemm_wmma<false, false><<<grid, 256, 0, stream>>>(
                tmid, w2b, b2p, h, nullptr,
                /*Mreal=*/N_NODES, /*Nreal=*/EMB, /*ldc=*/EMB,
                /*lda=*/N1_PAD, /*ldb=*/N2_PAD, /*Kpad=*/K2_PAD);
        }

        // BatchNorm (training-mode batch stats) + ReLU except last layer
        k_zero<<<nblk(EMB), 256, 0, stream>>>(bnSum, EMB);
        k_zero<<<nblk(EMB), 256, 0, stream>>>(bnSq, EMB);
        k_bn_stats<<<(N_NODES + 255) / 256, 256, 0, stream>>>(h, bnSum, bnSq, N_NODES);
        k_bn_apply<<<nblk(nodeElems), 256, 0, stream>>>(
            h, bnSum, bnSq, bn_gamma + (size_t)l * EMB, bn_beta + (size_t)l * EMB,
            (int)nodeElems, N_NODES, (l < N_LAYERS - 1) ? 1 : 0);
    }

    // global mean pool + prediction head
    k_zero<<<nblk((long long)N_GRAPHS * EMB), 256, 0, stream>>>(pool, N_GRAPHS * EMB);
    k_zero<<<nblk(N_GRAPHS), 256, 0, stream>>>(cnt, N_GRAPHS);
    k_pool_accum<<<nblk(nodeElems), 256, 0, stream>>>(h, batch, pool, (int)nodeElems);
    k_pool_count<<<nblk(N_NODES), 256, 0, stream>>>(batch, cnt, N_NODES);
    k_head<<<nblk((long long)N_GRAPHS * N_TASKS), 256, 0, stream>>>(
        pool, cnt, Wp, bp, out, N_GRAPHS * N_TASKS);
}
